// MQA_755914244506
// MI455X (gfx1250) — matrix operand order repaired and checked by reading, not compiled, not run on hardware
//
#include <hip/hip_runtime.h>
#include <hip/hip_bf16.h>

// ---------------------------------------------------------------------------
// MQA forward for MI455X (gfx1250, wave32, WMMA 16x16x32 f16 -> f32).
// B=2, T=2048, D=2048, H=16, HD=128.
// Pipeline: cast/transpose -> QKV WMMA GEMM (32x64 reg-blocked, K templated
// so subtile offsets fold into the 24-bit IOFFSET -> no pointer spills) ->
// flash-attn WMMA (TDM-staged K tiles) -> out WMMA GEMM
// ---------------------------------------------------------------------------

#define BATCH   2
#define SEQ     2048
#define DMODEL  2048
#define NHEADS  16
#define HDIM    128

typedef _Float16 half8  __attribute__((ext_vector_type(8)));
typedef _Float16 v16h_t __attribute__((ext_vector_type(16)));
typedef float    v8f_t  __attribute__((ext_vector_type(8)));
typedef unsigned int u32x4 __attribute__((ext_vector_type(4)));
typedef int          i32x4 __attribute__((ext_vector_type(4)));
typedef int          i32x8 __attribute__((ext_vector_type(8)));

union U16 { v16h_t v; half8 h[2]; };

#if defined(__has_builtin)
# if __has_builtin(__builtin_amdgcn_tensor_load_to_lds)
#  define HAVE_TDM 1
# endif
#endif

// ---------------------------------------------------------------- prologue --
__global__ void cast_f32_to_f16(const float* __restrict__ src,
                                _Float16* __restrict__ dst, size_t n) {
  size_t i = (size_t)blockIdx.x * blockDim.x + threadIdx.x;
  if (i < n) dst[i] = (_Float16)src[i];
}

// W: K x N (row major) -> Wt: N x K (row major), cast to f16.
__global__ void transpose_cast_f16(const float* __restrict__ W,
                                   _Float16* __restrict__ Wt,
                                   int K, int N) {
  size_t i = (size_t)blockIdx.x * blockDim.x + threadIdx.x;
  size_t total = (size_t)K * N;
  if (i < total) {
    int n = (int)(i % N);
    int k = (int)(i / N);
    Wt[(size_t)n * K + k] = (_Float16)W[i];
  }
}

// v_h: (B, T, HD) -> vt_h: (B, HD, T)
__global__ void transpose_v_f16(const _Float16* __restrict__ v_h,
                                _Float16* __restrict__ vt_h) {
  size_t i = (size_t)blockIdx.x * blockDim.x + threadIdx.x;
  size_t total = (size_t)BATCH * SEQ * HDIM;
  if (i < total) {
    int d = (int)(i % HDIM);
    int t = (int)((i / HDIM) % SEQ);
    int b = (int)(i / ((size_t)HDIM * SEQ));
    vt_h[((size_t)b * HDIM + d) * SEQ + t] = v_h[i];
  }
}

// -------------------------------------------------------------- WMMA GEMM --
// C[m,n] = sum_k A[m,k] * Bt[n,k] + bias[n]
// One wave per 32x64 output tile (2x4 WMMA subtiles): each A fragment is
// reused 4x and each B fragment 2x.  K is a template parameter so the
// mi*16*K / ni*16*K subtile strides become compile-time constants and fold
// into the global_load 24-bit immediate offset (2 base pointers total).
#define MBLK 2
#define NBLK 4
template <int K>
__global__ __launch_bounds__(32, 4)
void gemm_wmma_f16(const _Float16* __restrict__ A,
                   const _Float16* __restrict__ Bt,
                   const float* __restrict__ bias,
                   float* __restrict__ C32,       // optional
                   _Float16* __restrict__ C16,    // optional
                   int M, int N) {
  const int lane     = threadIdx.x & 31;
  const int lane16   = lane & 15;
  const int half_sel = lane >> 4;          // 0: lanes 0-15, 1: lanes 16-31
  const int m0 = blockIdx.y * (16 * MBLK);
  const int n0 = blockIdx.x * (16 * NBLK);

  const _Float16* __restrict__ abase = A  + (size_t)(m0 + lane16) * K;
  const _Float16* __restrict__ bbase = Bt + (size_t)(n0 + lane16) * K;

  v8f_t acc[MBLK][NBLK];
#pragma unroll
  for (int mi = 0; mi < MBLK; ++mi)
#pragma unroll
    for (int ni = 0; ni < NBLK; ++ni)
      acc[mi][ni] = (v8f_t){};

  for (int k0 = 0; k0 < K; k0 += 32) {
    __builtin_prefetch(abase + k0 + 256, 0, 1);   // global_prefetch_b8
    __builtin_prefetch(bbase + k0 + 256, 0, 1);
    U16 a[MBLK], b[NBLK];
#pragma unroll
    for (int mi = 0; mi < MBLK; ++mi) {
      // A fragment (16x32 f16): lane row = m-base+lane16, 16 contiguous K;
      // mi*16*K is a compile-time constant -> folds into IOFFSET
      a[mi].h[0] = *(const half8*)(abase + mi * 16 * K + k0 + half_sel * 16);
      a[mi].h[1] = *(const half8*)(abase + mi * 16 * K + k0 + half_sel * 16 + 8);
    }
#pragma unroll
    for (int ni = 0; ni < NBLK; ++ni) {
      // B fragment (32x16 f16): lane col = n-base+lane16, 16 contiguous K
      b[ni].h[0] = *(const half8*)(bbase + ni * 16 * K + k0 + half_sel * 16);
      b[ni].h[1] = *(const half8*)(bbase + ni * 16 * K + k0 + half_sel * 16 + 8);
    }
#pragma unroll
    for (int mi = 0; mi < MBLK; ++mi)
#pragma unroll
      for (int ni = 0; ni < NBLK; ++ni)
        acc[mi][ni] = __builtin_amdgcn_wmma_f32_16x16x32_f16(
            false, a[mi].v, false, b[ni].v, (short)0, acc[mi][ni],
            false, false);
  }

#pragma unroll
  for (int ni = 0; ni < NBLK; ++ni) {
    const float bb = bias ? bias[n0 + ni * 16 + lane16] : 0.0f;
#pragma unroll
    for (int mi = 0; mi < MBLK; ++mi) {
#pragma unroll
      for (int r = 0; r < 8; ++r) {
        const int m = m0 + mi * 16 + r + 8 * half_sel;
        const int n = n0 + ni * 16 + lane16;
        const float val = acc[mi][ni][r] + bb;
        if (C32) C32[(size_t)m * N + n] = val;
        if (C16) C16[(size_t)m * N + n] = (_Float16)val;
      }
    }
  }
}

// -------------------------------------------------------- flash attention --
// One wave per (b, h, 16-row Q tile).  Online softmax, causal mask.
// K tiles (32 x 128 f16) are staged into LDS by the Tensor Data Mover when
// available; B fragments then come from ds_load instead of global.
// q_h : (B, T, H*HD) f16      k_h : (B, T, HD) f16
// vt_h: (B, HD, T)  f16       ctx_h: (B, T, H*HD) f16
__global__ __launch_bounds__(32, 4)
void attn_wmma(const _Float16* __restrict__ q_h,
               const _Float16* __restrict__ k_h,
               const _Float16* __restrict__ vt_h,
               _Float16* __restrict__ ctx_h) {
  __shared__ __align__(16) _Float16 Ptile[16][32];
#ifdef HAVE_TDM
  __shared__ __align__(16) _Float16 Ktile[32 * HDIM];   // 8 KB TDM staging
#endif

  const int lane     = threadIdx.x & 31;
  const int lane16   = lane & 15;
  const int half_sel = lane >> 4;

  int tile = blockIdx.x;
  const int nTt = SEQ / 16;
  const int tt = tile % nTt; tile /= nTt;
  const int h  = tile % NHEADS;
  const int b  = tile / NHEADS;
  const int t0 = tt * 16;
  const float scale = 0.08838834764831845f;   // 1/sqrt(128)

  // Q A-fragments: head dim 128 = 4 K-steps of 32
  U16 aq[4];
  const _Float16* __restrict__ qrow =
      q_h + (size_t)(b * SEQ + t0 + lane16) * (NHEADS * HDIM) + h * HDIM;
#pragma unroll
  for (int kk = 0; kk < 4; ++kk) {
    aq[kk].h[0] = *(const half8*)(qrow + kk * 32 + half_sel * 16);
    aq[kk].h[1] = *(const half8*)(qrow + kk * 32 + half_sel * 16 + 8);
  }

  v8f_t cacc[8];
#pragma unroll
  for (int i = 0; i < 8; ++i) cacc[i] = (v8f_t){};

  float rmax[8], rsum[8];
#pragma unroll
  for (int r = 0; r < 8; ++r) { rmax[r] = -1e30f; rsum[r] = 0.0f; }

  const _Float16* __restrict__ kb  = k_h  + (size_t)b * SEQ * HDIM;
  const _Float16* __restrict__ vtb = vt_h + (size_t)b * HDIM * SEQ;

  const int send = t0 + 16;                    // causal upper bound (exclusive)
  for (int s0 = 0; s0 < send; s0 += 32) {
#ifdef HAVE_TDM
    {
      // Make sure previous iteration's ds reads finished before DMA overwrite
      asm volatile("s_wait_dscnt 0" ::: "memory");
      const unsigned long long ga =
          (unsigned long long)(size_t)(kb + (size_t)s0 * HDIM);
      const unsigned ldsOff = (unsigned)(size_t)(void*)Ktile;
      // D# group 0: count=1 | lds_addr | global_addr | type=2 (ISA 8.3)
      u32x4 g0;
      g0[0] = 1u;
      g0[1] = ldsOff;
      g0[2] = (unsigned)(ga & 0xFFFFFFFFull);
      g0[3] = (unsigned)((ga >> 32) & 0x01FFFFFFull) | (2u << 30);
      // D# group 1: data_size=2B, tensor 128 x SEQ, tile 128 x 32, stride 128
      const unsigned td0 = HDIM, td1 = SEQ;
      i32x8 g1;
      g1[0] = (int)(1u << 16);                              // data_size=1 (2B)
      g1[1] = (int)((td0 & 0xFFFFu) << 16);                 // tensor_dim0 lo
      g1[2] = (int)((td0 >> 16) | ((td1 & 0xFFFFu) << 16)); // dim0 hi|dim1 lo
      g1[3] = (int)((td1 >> 16) | (128u << 16));            // dim1 hi|tile0=128
      g1[4] = (int)32u;                                     // tile_dim1=32
      g1[5] = (int)HDIM;                                    // dim0_stride=128
      g1[6] = 0;
      g1[7] = 0;
      const i32x4 gz = {0, 0, 0, 0};
# if __has_include(<hip/amd_detail/amd_gfx1250_TDM.h>)
      const i32x8 gz8 = {0, 0, 0, 0, 0, 0, 0, 0};
      __builtin_amdgcn_tensor_load_to_lds(g0, g1, gz, gz, gz8, 0);
# else
      __builtin_amdgcn_tensor_load_to_lds(g0, g1, gz, gz, 0);
# endif
      asm volatile("s_wait_tensorcnt 0" ::: "memory");
    }
#endif
#pragma unroll
    for (int sub = 0; sub < 2; ++sub) {
      const int sc0 = s0 + sub * 16;
      // scores: S = Q @ K^T for 16x16 tile, K-dim = head dim (4 WMMAs)
      v8f_t sacc = (v8f_t){};
#ifdef HAVE_TDM
      const _Float16* __restrict__ krow =
          Ktile + (size_t)(sub * 16 + lane16) * HDIM;
#else
      const _Float16* __restrict__ krow = kb + (size_t)(sc0 + lane16) * HDIM;
#endif
#pragma unroll
      for (int kk = 0; kk < 4; ++kk) {
        U16 bkf;
        bkf.h[0] = *(const half8*)(krow + kk * 32 + half_sel * 16);
        bkf.h[1] = *(const half8*)(krow + kk * 32 + half_sel * 16 + 8);
        sacc = __builtin_amdgcn_wmma_f32_16x16x32_f16(
            false, aq[kk].v, false, bkf.v, (short)0, sacc, false, false);
      }
      // mask + scale + online softmax update
#pragma unroll
      for (int r = 0; r < 8; ++r) {
        const int trow = t0 + r + 8 * half_sel;
        const int scol = sc0 + lane16;
        float val = sacc[r] * scale;
        if (scol > trow) val = -1e30f;
        float tmax = val;
#pragma unroll
        for (int off = 1; off < 16; off <<= 1)
          tmax = fmaxf(tmax, __shfl_xor(tmax, off, 32));
        const float nmax = fmaxf(rmax[r], tmax);
        const float corr = __expf(rmax[r] - nmax);
        const float p    = __expf(val - nmax);
        float psum = p;
#pragma unroll
        for (int off = 1; off < 16; off <<= 1)
          psum += __shfl_xor(psum, off, 32);
        rsum[r] = rsum[r] * corr + psum;
        rmax[r] = nmax;
#pragma unroll
        for (int dt = 0; dt < 8; ++dt) cacc[dt][r] *= corr;
        Ptile[r + 8 * half_sel][sub * 16 + lane16] = (_Float16)p;
      }
    }

    // LDS P stores -> loads within one wave; DS ops in order, wait to be safe
    asm volatile("s_wait_dscnt 0" ::: "memory");

    // reload P as a 16x32 A-fragment
    U16 ap;
    ap.h[0] = *(const half8*)(&Ptile[lane16][half_sel * 16]);
    ap.h[1] = *(const half8*)(&Ptile[lane16][half_sel * 16 + 8]);

    // ctx += P @ V  (8 d-tiles of 16, K-dim = 32 seq positions)
#pragma unroll
    for (int dt = 0; dt < 8; ++dt) {
      U16 bvf;
      const _Float16* __restrict__ vcol =
          vtb + (size_t)(dt * 16 + lane16) * SEQ + s0 + half_sel * 16;
      bvf.h[0] = *(const half8*)(vcol);
      bvf.h[1] = *(const half8*)(vcol + 8);
      cacc[dt] = __builtin_amdgcn_wmma_f32_16x16x32_f16(
          false, ap.v, false, bvf.v, (short)0, cacc[dt], false, false);
    }
  }

  // normalize and store ctx as f16
#pragma unroll
  for (int r = 0; r < 8; ++r) {
    const float inv = 1.0f / rsum[r];
    const int trow = t0 + r + 8 * half_sel;
    _Float16* __restrict__ crow =
        ctx_h + (size_t)(b * SEQ + trow) * (NHEADS * HDIM) + h * HDIM;
#pragma unroll
    for (int dt = 0; dt < 8; ++dt)
      crow[dt * 16 + lane16] = (_Float16)(cacc[dt][r] * inv);
  }
}

// ------------------------------------------------------------------- host --
extern "C" void kernel_launch(void* const* d_in, const int* in_sizes, int n_in,
                              void* d_out, int out_size, void* d_ws, size_t ws_size,
                              hipStream_t stream) {
  (void)in_sizes; (void)n_in; (void)out_size; (void)ws_size;

  const float* x  = (const float*)d_in[0];
  /* d_in[1] = start_pos, always 0 for this shape */
  const float* Wq = (const float*)d_in[2];
  const float* bq = (const float*)d_in[3];
  const float* Wk = (const float*)d_in[4];
  const float* bk = (const float*)d_in[5];
  const float* Wv = (const float*)d_in[6];
  const float* bv = (const float*)d_in[7];
  const float* Wo = (const float*)d_in[8];
  const float* bo = (const float*)d_in[9];

  float* out  = (float*)d_out;                              // (B,T,D)
  float* kcat = out  + (size_t)BATCH * SEQ * DMODEL;        // (B,1,T,HD)
  float* vcat = kcat + (size_t)BATCH * SEQ * HDIM;          // (B,1,T,HD)

  // workspace carve-up (f16 buffers)
  char* w = (char*)d_ws;
  const size_t n_x   = (size_t)BATCH * SEQ * DMODEL;        // 8.4M
  const size_t n_kv  = (size_t)BATCH * SEQ * HDIM;          // 0.52M
  _Float16* x_h   = (_Float16*)w;               w += n_x * 2;
  _Float16* q_h   = (_Float16*)w;               w += n_x * 2;
  _Float16* ctx_h = (_Float16*)w;               w += n_x * 2;
  _Float16* Wqt_h = (_Float16*)w;               w += (size_t)DMODEL * DMODEL * 2;
  _Float16* Wot_h = (_Float16*)w;               w += (size_t)DMODEL * DMODEL * 2;
  _Float16* Wkt_h = (_Float16*)w;               w += (size_t)DMODEL * HDIM * 2;
  _Float16* Wvt_h = (_Float16*)w;               w += (size_t)DMODEL * HDIM * 2;
  _Float16* k_h   = (_Float16*)w;               w += n_kv * 2;
  _Float16* v_h   = (_Float16*)w;               w += n_kv * 2;
  _Float16* vt_h  = (_Float16*)w;               w += n_kv * 2;

  const int TPB = 256;
  // 1) cast x to f16
  cast_f32_to_f16<<<dim3((unsigned)((n_x + TPB - 1) / TPB)), TPB, 0, stream>>>(x, x_h, n_x);
  // 2) transpose+cast weights
  {
    size_t nw = (size_t)DMODEL * DMODEL;
    transpose_cast_f16<<<dim3((unsigned)((nw + TPB - 1) / TPB)), TPB, 0, stream>>>(Wq, Wqt_h, DMODEL, DMODEL);
    transpose_cast_f16<<<dim3((unsigned)((nw + TPB - 1) / TPB)), TPB, 0, stream>>>(Wo, Wot_h, DMODEL, DMODEL);
    size_t nkv = (size_t)DMODEL * HDIM;
    transpose_cast_f16<<<dim3((unsigned)((nkv + TPB - 1) / TPB)), TPB, 0, stream>>>(Wk, Wkt_h, DMODEL, HDIM);
    transpose_cast_f16<<<dim3((unsigned)((nkv + TPB - 1) / TPB)), TPB, 0, stream>>>(Wv, Wvt_h, DMODEL, HDIM);
  }

  const int M = BATCH * SEQ;   // 4096
  // 3) q = x @ Wq + bq   (f16 result for attention)
  gemm_wmma_f16<DMODEL><<<dim3(DMODEL / (16 * NBLK), M / (16 * MBLK)), 32, 0, stream>>>(
      x_h, Wqt_h, bq, (float*)nullptr, q_h, M, DMODEL);
  // 4) k = x @ Wk + bk   (f16 for attention + f32 into k_cat output)
  gemm_wmma_f16<DMODEL><<<dim3(HDIM / (16 * NBLK), M / (16 * MBLK)), 32, 0, stream>>>(
      x_h, Wkt_h, bk, kcat, k_h, M, HDIM);
  // 5) v = x @ Wv + bv   (f16 + f32 into v_cat output)
  gemm_wmma_f16<DMODEL><<<dim3(HDIM / (16 * NBLK), M / (16 * MBLK)), 32, 0, stream>>>(
      x_h, Wvt_h, bv, vcat, v_h, M, HDIM);
  // 6) transpose V to (B, HD, T) for contiguous WMMA B-fragments
  transpose_v_f16<<<dim3((unsigned)((n_kv + TPB - 1) / TPB)), TPB, 0, stream>>>(v_h, vt_h);
  // 7) flash attention
  attn_wmma<<<dim3(BATCH * NHEADS * (SEQ / 16)), 32, 0, stream>>>(q_h, k_h, vt_h, ctx_h);
  // 8) out = ctx @ Wo + bo
  gemm_wmma_f16<DMODEL><<<dim3(DMODEL / (16 * NBLK), M / (16 * MBLK)), 32, 0, stream>>>(
      ctx_h, Wot_h, bo, out, (_Float16*)nullptr, M, DMODEL);
}
